// RegionProposalNetwork_59184649339007
// MI455X (gfx1250) — matrix operand order repaired and checked by reading, not compiled, not run on hardware
//
#include <hip/hip_runtime.h>
#include <hip/hip_bf16.h>
#include <stdint.h>

typedef __attribute__((ext_vector_type(16))) __bf16    v16bf;
typedef __attribute__((ext_vector_type(8)))  float     v8f;
typedef __attribute__((ext_vector_type(4)))  unsigned  u32x4;
typedef __attribute__((ext_vector_type(2)))  unsigned  u32x2;
typedef __attribute__((ext_vector_type(4)))  float     f32x4;

union Frag16 { u32x4 q[2]; v16bf v; };

constexpr int Bsz  = 8, C = 1024, H = 50, W = 84;
constexpr int HW   = H * W;          // 4200
constexpr int NTOT = Bsz * HW;       // 33600 spatial positions (b,y,x)
constexpr int CO1  = 512;            // trunk output channels
constexpr int KTOT = C * 9;          // 9216 im2col K
constexpr int OBJ_C = 18, TRF_C = 36, OC = 54;
constexpr int MT = 64, NT2 = 256, KC = 32;         // trunk block tile
constexpr int NBLK_T = (NTOT + NT2 - 1) / NT2;     // 132
constexpr int NTH = 128;                           // heads N tile
constexpr int NBLK_H = (NTOT + NTH - 1) / NTH;     // 263
constexpr int APAD = 40;             // padded LDS row stride (bf16), 80B = 5x16B

static __device__ __forceinline__ unsigned short f32_to_bf16(float f) {
  unsigned u = __float_as_uint(f);
  u += 0x7FFFu + ((u >> 16) & 1u);   // round-to-nearest-even
  return (unsigned short)(u >> 16);
}

// CDNA5 async global->LDS copy (ASYNCcnt-tracked), per CDNA5 ISA
static __device__ __forceinline__ void async_copy_b128(unsigned lds_addr, const void* gaddr) {
  asm volatile("global_load_async_to_lds_b128 %0, %1, off"
               :: "v"(lds_addr), "v"(gaddr) : "memory");
}
// Split workgroup barrier: drain LDS stores + async A-copy, then signal arrival.
static __device__ __forceinline__ void barrier_signal_drained() {
  asm volatile("s_wait_dscnt 0x0\n\t"
               "s_wait_asynccnt 0x0\n\t"
               "s_barrier_signal -1" ::: "memory");
}
static __device__ __forceinline__ void barrier_wait() {
  asm volatile("s_barrier_wait -1" ::: "memory");
}

// ------------- Kernel 0a: repack w1 [512][1024][3][3] f32 -> bf16 [512][r*1024+c]
__global__ __launch_bounds__(256) void k_pack_w1(const float* __restrict__ w1,
                                                 unsigned short* __restrict__ wp) {
  int idx = blockIdx.x * 256 + threadIdx.x;
  if (idx >= CO1 * KTOT) return;
  int m = idx / KTOT;
  int k = idx - m * KTOT;        // destination k = r*1024 + c
  int r = k >> 10;
  int c = k & 1023;
  wp[idx] = f32_to_bf16(w1[m * KTOT + c * 9 + r]);
}

// ------------- Kernel 0b: features f32 -> bf16 (same layout), 4 elems/thread
__global__ __launch_bounds__(256) void k_pack_feat(const float* __restrict__ f,
                                                   unsigned short* __restrict__ o) {
  int i = blockIdx.x * 256 + threadIdx.x;            // quad index
  if (i >= (Bsz * C * HW) / 4) return;
  f32x4 v = reinterpret_cast<const f32x4*>(f)[i];
  u32x2 p;
  p.x = (unsigned)f32_to_bf16(v.x) | ((unsigned)f32_to_bf16(v.y) << 16);
  p.y = (unsigned)f32_to_bf16(v.z) | ((unsigned)f32_to_bf16(v.w) << 16);
  reinterpret_cast<u32x2*>(o)[i] = p;
}

// ------------- Kernel 0c: fuse obj(18)+trf(36) head weights -> bf16 [64][512], zero pad
__global__ __launch_bounds__(256) void k_pack_headw(const float* __restrict__ wobj,
                                                    const float* __restrict__ wtrf,
                                                    unsigned short* __restrict__ wc) {
  int idx = blockIdx.x * 256 + threadIdx.x;
  if (idx >= 64 * CO1) return;
  int o = idx >> 9, k = idx & 511;
  float f = 0.0f;
  if (o < OBJ_C)   f = wobj[o * CO1 + k];
  else if (o < OC) f = wtrf[(o - OBJ_C) * CO1 + k];
  wc[idx] = f32_to_bf16(f);
}

// ------------- Kernel 1: trunk 3x3 conv, implicit GEMM, double-buffered pipeline
// Block = 256 thr = 8 waves as 2(M)x4(N); wave tile 32(M)x64(N) -> 8 WMMA/chunk
__global__ __launch_bounds__(256) void k_trunk(const unsigned short* __restrict__ featb,
                                               const unsigned short* __restrict__ wp,
                                               const float* __restrict__ b1,
                                               unsigned short* __restrict__ hout) {
  __shared__ unsigned short sA[2][MT * APAD];    // [m][k] chunk, 64x32 padded
  __shared__ unsigned short sB[2][NT2 * APAD];   // [n][k] chunk, 256x32 padded

  const int tid  = threadIdx.x;
  const int wid  = tid >> 5, lane = tid & 31;
  const int half = lane >> 4, l15 = lane & 15;
  const int wm   = wid & 1;           // 2 M groups of 32
  const int wn   = wid >> 1;          // 4 N groups of 64
  const int m0   = blockIdx.y * MT;
  const int n0   = blockIdx.x * NT2;

  // Per-thread im2col column: thread tid feeds column nj = tid (all 32 k rows)
  const int nj = tid;
  const int n  = n0 + nj;
  const bool nvalid = n < NTOT;
  const int nc  = nvalid ? n : 0;
  const int bb  = nc / HW;
  const int rem = nc - bb * HW;
  const int y   = rem / W;
  const int x   = rem - y * W;

  unsigned short br[KC];              // staged B column (registers)
  unsigned brmask;                    // 0xFFFFFFFF if in-bounds else 0 (halo zero-pad)

  // Branchless gather: clamp address so loads always issue (one clause, no exec
  // divergence); zero-padding applied via mask when packing into LDS.
  auto load_B = [&](int kk) {
    const int r  = kk >> 5;
    const int c0 = (kk & 31) << 5;
    const int dy = r / 3 - 1;
    const int dx = r - (r / 3) * 3 - 1;
    const int yy = y + dy, xx = x + dx;
    const bool valid = nvalid && (unsigned)yy < (unsigned)H && (unsigned)xx < (unsigned)W;
    brmask = valid ? 0xFFFFFFFFu : 0u;
    const int yyc = valid ? yy : 0;
    const int xxc = valid ? xx : 0;
    const unsigned short* fsrc = featb + (((size_t)bb * C + c0) * H + yyc) * W + xxc;
#pragma unroll
    for (int i = 0; i < KC; ++i)
      br[i] = fsrc[(size_t)i * HW];
  };

  auto store_B = [&](int buf) {
    unsigned* dst = reinterpret_cast<unsigned*>(&sB[buf][nj * APAD]);
#pragma unroll
    for (int i = 0; i < KC / 2; ++i)
      dst[i] = ((unsigned)br[2 * i] | ((unsigned)br[2 * i + 1] << 16)) & brmask;
  };

  auto stage_A = [&](int kk, int buf) {            // async global->LDS, bf16 copy
    const int r  = kk >> 5;
    const int c0 = (kk & 31) << 5;
    const int mi = tid >> 2, s = tid & 3;
    const void* g = wp + (size_t)(m0 + mi) * KTOT + (r << 10) + c0 + s * 8;
    unsigned lds = (unsigned)(uintptr_t)&sA[buf][mi * APAD + s * 8];
    async_copy_b128(lds, g);
  };

  v8f acc[8];
#pragma unroll
  for (int i = 0; i < 8; ++i) acc[i] = (v8f)0.0f;

  // ---- pipeline prologue
  load_B(0);
  stage_A(0, 0);

  for (int kk = 0; kk < KTOT / KC; ++kk) {         // 288 chunks
    const int cur = kk & 1;
    const bool more = (kk + 1) < (KTOT / KC);
    store_B(cur);                                  // B(kk) regs -> LDS
    barrier_signal_drained();                      // dscnt+asynccnt drained, arrive
    if (more) load_B(kk + 1);                      // global->VGPR: safe pre-wait
    barrier_wait();
    if (more) stage_A(kk + 1, cur ^ 1);            // async LDS write: must be post-wait

    Frag16 a0, a1;
    const unsigned short* ar0 = &sA[cur][(wm * 32 + l15) * APAD];
    const unsigned short* ar1 = &sA[cur][(wm * 32 + 16 + l15) * APAD];
    a0.q[0] = *reinterpret_cast<const u32x4*>(ar0 + half * 8);
    a0.q[1] = *reinterpret_cast<const u32x4*>(ar0 + 16 + half * 8);
    a1.q[0] = *reinterpret_cast<const u32x4*>(ar1 + half * 8);
    a1.q[1] = *reinterpret_cast<const u32x4*>(ar1 + 16 + half * 8);
#pragma unroll
    for (int no = 0; no < 4; ++no) {
      Frag16 b;
      const unsigned short* brow = &sB[cur][(wn * 64 + no * 16 + l15) * APAD];
      b.q[0] = *reinterpret_cast<const u32x4*>(brow + half * 8);
      b.q[1] = *reinterpret_cast<const u32x4*>(brow + 16 + half * 8);
      acc[no]     = __builtin_amdgcn_wmma_f32_16x16x32_bf16(
          false, a0.v, false, b.v, (short)0, acc[no],     false, false);
      acc[4 + no] = __builtin_amdgcn_wmma_f32_16x16x32_bf16(
          false, a1.v, false, b.v, (short)0, acc[4 + no], false, false);
    }
  }

  // ---- epilogue: +bias, ReLU, store h as bf16 [n][512]
#pragma unroll
  for (int no = 0; no < 4; ++no) {
    const int nn = n0 + wn * 64 + no * 16 + l15;
    if (nn < NTOT) {
#pragma unroll
      for (int ma = 0; ma < 2; ++ma) {
#pragma unroll
        for (int v = 0; v < 8; ++v) {
          const int m = m0 + wm * 32 + ma * 16 + half * 8 + v;
          float t = acc[ma * 4 + no][v] + b1[m];
          t = t > 0.0f ? t : 0.0f;
          hout[(size_t)nn * CO1 + m] = f32_to_bf16(t);
        }
      }
    }
  }
}

// ------------- Kernel 2: fused 1x1 heads; A and B frags straight from global bf16
__global__ __launch_bounds__(256) void k_heads(const unsigned short* __restrict__ h,
                                               const unsigned short* __restrict__ wc,
                                               const float* __restrict__ bobj,
                                               const float* __restrict__ btrf,
                                               float* __restrict__ out) {
  const int tid  = threadIdx.x;
  const int wid  = tid >> 5, lane = tid & 31;
  const int half = lane >> 4, l15 = lane & 15;
  const int n0   = blockIdx.x * NTH;

  const int nrow   = n0 + wid * 16 + l15;
  const int nclamp = nrow < NTOT ? nrow : NTOT - 1;
  const u32x4* hrow = reinterpret_cast<const u32x4*>(h + (size_t)nclamp * CO1);

  v8f acc[4];
#pragma unroll
  for (int i = 0; i < 4; ++i) acc[i] = (v8f)0.0f;

  for (int kc = 0; kc < CO1 / KC; ++kc) {          // 16 chunks
    Frag16 a;
    a.q[0] = hrow[kc * 4 + half];
    a.q[1] = hrow[kc * 4 + 2 + half];
#pragma unroll
    for (int ot = 0; ot < 4; ++ot) {
      Frag16 b;
      const u32x4* wrow = reinterpret_cast<const u32x4*>(wc + (size_t)(ot * 16 + l15) * CO1);
      b.q[0] = wrow[kc * 4 + half];
      b.q[1] = wrow[kc * 4 + 2 + half];
      acc[ot] = __builtin_amdgcn_wmma_f32_16x16x32_bf16(
          false, a.v, false, b.v, (short)0, acc[ot], false, false);
    }
  }

  // bias + permute(0,2,3,1).view scatter into concatenated (obj, trf)
#pragma unroll
  for (int ot = 0; ot < 4; ++ot) {
#pragma unroll
    for (int v = 0; v < 8; ++v) {
      const int nn = n0 + wid * 16 + half * 8 + v;
      const int o  = ot * 16 + l15;
      if (nn < NTOT && o < OC) {
        const int b   = nn / HW;
        const int rem = nn - b * HW;               // y*W + x
        if (o < OBJ_C) {
          const int a9 = o >> 1, j = o & 1;
          out[(size_t)b * (HW * OBJ_C) + (size_t)(rem * 9 + a9) * 2 + j] =
              acc[ot][v] + bobj[o];
        } else {
          const int oo = o - OBJ_C;
          const int a9 = oo >> 2, j = oo & 3;
          out[(size_t)NTOT * OBJ_C + (size_t)b * (HW * TRF_C) +
              (size_t)(rem * 9 + a9) * 4 + j] = acc[ot][v] + btrf[oo];
        }
      }
    }
  }
}

extern "C" void kernel_launch(void* const* d_in, const int* in_sizes, int n_in,
                              void* d_out, int out_size, void* d_ws, size_t ws_size,
                              hipStream_t stream) {
  const float* feat = (const float*)d_in[0];
  const float* w1   = (const float*)d_in[1];
  const float* b1   = (const float*)d_in[2];
  const float* wobj = (const float*)d_in[3];
  const float* bobj = (const float*)d_in[4];
  const float* wtrf = (const float*)d_in[5];
  const float* btrf = (const float*)d_in[6];
  float* out = (float*)d_out;

  // workspace layout (bf16 halves): wp | featb | hbuf | wcomb   (~113 MB)
  unsigned short* wp    = (unsigned short*)d_ws;
  unsigned short* featb = wp + (size_t)CO1 * KTOT;             // 512*9216
  unsigned short* hbuf  = featb + (size_t)Bsz * C * HW;        // 8*1024*4200
  unsigned short* wcomb = hbuf + (size_t)NTOT * CO1;           // 33600*512

  k_pack_w1<<<(CO1 * KTOT + 255) / 256, 256, 0, stream>>>(w1, wp);
  k_pack_feat<<<(Bsz * C * HW / 4 + 255) / 256, 256, 0, stream>>>(feat, featb);
  k_pack_headw<<<(64 * CO1 + 255) / 256, 256, 0, stream>>>(wobj, wtrf, wcomb);

  dim3 g1(NBLK_T, CO1 / MT);                                   // 132 x 8
  k_trunk<<<g1, 256, 0, stream>>>(featb, wp, b1, hbuf);
  k_heads<<<NBLK_H, 256, 0, stream>>>(hbuf, wcomb, bobj, btrf, out);
}
